// SparseMoE_15281493639607
// MI455X (gfx1250) — matrix operand order repaired and checked by reading, not compiled, not run on hardware
//
#include <hip/hip_runtime.h>
#include <hip/hip_bf16.h>
#include <math.h>

#define HIDDEN 1024
#define INTER  2816
#define NEXP   8
#define TM     64
#define TN     256

typedef _Float16 v16h __attribute__((ext_vector_type(16)));
typedef _Float16 v8h  __attribute__((ext_vector_type(8)));
typedef float    v8f  __attribute__((ext_vector_type(8)));

// ---------------- init: zero expert counts, poison pair slots ----------------
__global__ void moe_init_kernel(int* __restrict__ ecount, int* __restrict__ ptok,
                                float* __restrict__ pwt, int pmax) {
  int i = blockIdx.x * blockDim.x + threadIdx.x;
  if (i < NEXP) ecount[i] = 0;
  if (i < pmax) { ptok[i] = -1; pwt[i] = 0.f; }
}

// ---------------- gating: one wave32 per token ----------------
__global__ __launch_bounds__(256)
void moe_gate_kernel(const float* __restrict__ x, const float* __restrict__ gw,
                     int* __restrict__ t2i, float* __restrict__ t2w,
                     int* __restrict__ ecount, int T) {
  const int wv = threadIdx.x >> 5, lane = threadIdx.x & 31;
  const int t = blockIdx.x * 8 + wv;
  if (t >= T) return;
  const float* xr = x + (size_t)t * HIDDEN;
  float acc[NEXP];
#pragma unroll
  for (int e = 0; e < NEXP; ++e) acc[e] = 0.f;
  for (int h = lane; h < HIDDEN; h += 32) {
    float xv = xr[h];
#pragma unroll
    for (int e = 0; e < NEXP; ++e) acc[e] += xv * gw[e * HIDDEN + h];
  }
#pragma unroll
  for (int e = 0; e < NEXP; ++e)
#pragma unroll
    for (int off = 16; off; off >>= 1)
      acc[e] += __shfl_xor(acc[e], off, 32);
  if (lane == 0) {
    // top-1 (earliest index on ties, like jax.lax.top_k)
    int e0 = 0;
#pragma unroll
    for (int e = 1; e < NEXP; ++e) if (acc[e] > acc[e0]) e0 = e;
    // top-2
    int e1 = (e0 == 0) ? 1 : 0;
#pragma unroll
    for (int e = 0; e < NEXP; ++e) if (e != e0 && acc[e] > acc[e1]) e1 = e;
    // softmax -> renormalized pair weights: w0 = 1/(1+exp(l1-l0))
    float w0 = 1.f / (1.f + expf(acc[e1] - acc[e0]));
    float w1v = 1.f - w0;
    t2i[2 * t] = e0; t2i[2 * t + 1] = e1;
    t2w[2 * t] = w0; t2w[2 * t + 1] = w1v;
    atomicAdd(&ecount[e0], 1);
    atomicAdd(&ecount[e1], 1);
  }
}

// ---------------- scan: TM-padded expert offsets ----------------
__global__ void moe_scan_kernel(const int* __restrict__ ecount,
                                int* __restrict__ poff, int* __restrict__ ecursor) {
  if (threadIdx.x == 0 && blockIdx.x == 0) {
    int off = 0;
    for (int e = 0; e < NEXP; ++e) {
      poff[e] = off;
      ecursor[e] = off;
      off += (ecount[e] + TM - 1) & ~(TM - 1);
    }
    poff[NEXP] = off;
  }
}

// ---------------- scatter tokens into expert-sorted pair slots ----------------
__global__ void moe_scatter_kernel(const int* __restrict__ t2i, const float* __restrict__ t2w,
                                   int* __restrict__ ecursor, int* __restrict__ ptok,
                                   float* __restrict__ pwt, int* __restrict__ posmap, int T) {
  int t = blockIdx.x * blockDim.x + threadIdx.x;
  if (t >= T) return;
#pragma unroll
  for (int k = 0; k < 2; ++k) {
    int e = t2i[2 * t + k];
    int pos = atomicAdd(&ecursor[e], 1);
    ptok[pos] = t;
    pwt[pos] = t2w[2 * t + k];
    posmap[2 * t + k] = pos;
  }
}

// ---------------- GEMM1: act = silu(x@Wg) * (x@Wu), f16 out ----------------
__global__ __launch_bounds__(256)
void moe_gemm1_kernel(const float* __restrict__ x, const float* __restrict__ w1,
                      const int* __restrict__ ptok, const int* __restrict__ poff,
                      _Float16* __restrict__ act) {
  __shared__ __align__(16) _Float16 As[TM * 32];   // [row][k]
  __shared__ __align__(16) _Float16 Bg[TN * 32];   // [n][k] (transposed)
  __shared__ __align__(16) _Float16 Bu[TN * 32];
  __shared__ int toks[TM];

  const int m0 = blockIdx.x * TM;
  if (m0 >= poff[NEXP]) return;
  int e = 0;
  while (e < NEXP - 1 && m0 >= poff[e + 1]) ++e;
  const int n0 = blockIdx.y * TN;
  const int tid = threadIdx.x;
  const int wv = tid >> 5, lane = tid & 31;
  const int lh = lane >> 4, l15 = lane & 15;

  if (tid < TM) toks[tid] = ptok[m0 + tid];
  __syncthreads();

  const float* w1e = w1 + (size_t)e * HIDDEN * (2 * INTER);

  v8f zero = {};
  v8f accg[4][2], accu[4][2];
#pragma unroll
  for (int i = 0; i < 4; ++i)
#pragma unroll
    for (int j = 0; j < 2; ++j) { accg[i][j] = zero; accu[i][j] = zero; }

  for (int k0 = 0; k0 < HIDDEN; k0 += 32) {
    // A tile: gather token rows, f32->f16
#pragma unroll
    for (int i = 0; i < 8; ++i) {
      int idx = tid * 8 + i;
      int r = idx >> 5, kk = idx & 31;
      int tok = toks[r];
      As[idx] = (tok >= 0) ? (_Float16)x[(size_t)tok * HIDDEN + k0 + kk] : (_Float16)0.f;
    }
    // B tiles: thread owns output column `tid`; coalesced across lanes per kk
    {
      const float* cg = w1e + (n0 + tid);
      const float* cu = w1e + (INTER + n0 + tid);
#pragma unroll 4
      for (int kk = 0; kk < 32; ++kk) {
        Bg[tid * 32 + kk] = (_Float16)cg[(size_t)(k0 + kk) * (2 * INTER)];
        Bu[tid * 32 + kk] = (_Float16)cu[(size_t)(k0 + kk) * (2 * INTER)];
      }
      // prefetch next K-chunk of the strided weight columns
      if (k0 + 32 < HIDDEN) {
#pragma unroll 4
        for (int kk = 0; kk < 32; kk += 8) {
          __builtin_prefetch(&cg[(size_t)(k0 + 32 + kk) * (2 * INTER)], 0, 1);
          __builtin_prefetch(&cu[(size_t)(k0 + 32 + kk) * (2 * INTER)], 0, 1);
        }
      }
    }
    __syncthreads();
#pragma unroll
    for (int mi = 0; mi < 4; ++mi) {
      const int row = mi * 16 + l15;
      v8h alo = *(const v8h*)&As[row * 32 + lh * 16];
      v8h ahi = *(const v8h*)&As[row * 32 + lh * 16 + 8];
      v16h a = __builtin_shufflevector(alo, ahi, 0,1,2,3,4,5,6,7,8,9,10,11,12,13,14,15);
#pragma unroll
      for (int ni = 0; ni < 2; ++ni) {
        const int col = wv * 32 + ni * 16 + l15;
        v8h gl = *(const v8h*)&Bg[col * 32 + lh * 16];
        v8h gh = *(const v8h*)&Bg[col * 32 + lh * 16 + 8];
        v16h bg = __builtin_shufflevector(gl, gh, 0,1,2,3,4,5,6,7,8,9,10,11,12,13,14,15);
        accg[mi][ni] = __builtin_amdgcn_wmma_f32_16x16x32_f16(
            false, a, false, bg, (short)0, accg[mi][ni], false, false);
        v8h ul = *(const v8h*)&Bu[col * 32 + lh * 16];
        v8h uh = *(const v8h*)&Bu[col * 32 + lh * 16 + 8];
        v16h bu = __builtin_shufflevector(ul, uh, 0,1,2,3,4,5,6,7,8,9,10,11,12,13,14,15);
        accu[mi][ni] = __builtin_amdgcn_wmma_f32_16x16x32_f16(
            false, a, false, bu, (short)0, accu[mi][ni], false, false);
      }
    }
    __syncthreads();
  }

  // epilogue: SiLU(gate)*up -> act (f16)
#pragma unroll
  for (int mi = 0; mi < 4; ++mi)
#pragma unroll
    for (int ni = 0; ni < 2; ++ni)
#pragma unroll
      for (int r = 0; r < 8; ++r) {
        int row = m0 + mi * 16 + lh * 8 + r;
        int col = n0 + wv * 32 + ni * 16 + l15;
        float g = accg[mi][ni][r];
        float u = accu[mi][ni][r];
        float a = (g / (1.f + expf(-g))) * u;
        act[(size_t)row * INTER + col] = (_Float16)a;
      }
}

// ---------------- GEMM2: outp = (act @ w2[e]) * routing_weight ----------------
__global__ __launch_bounds__(256)
void moe_gemm2_kernel(const _Float16* __restrict__ act, const float* __restrict__ w2,
                      const int* __restrict__ poff, const float* __restrict__ pwt,
                      float* __restrict__ outp) {
  __shared__ __align__(16) _Float16 As[TM * 32];
  __shared__ __align__(16) _Float16 Bs[TN * 32];   // [n][k]

  const int m0 = blockIdx.x * TM;
  if (m0 >= poff[NEXP]) return;
  int e = 0;
  while (e < NEXP - 1 && m0 >= poff[e + 1]) ++e;
  const int n0 = blockIdx.y * TN;
  const int tid = threadIdx.x;
  const int wv = tid >> 5, lane = tid & 31;
  const int lh = lane >> 4, l15 = lane & 15;

  const float* w2e = w2 + (size_t)e * INTER * HIDDEN;

  v8f zero = {};
  v8f acc[4][2];
#pragma unroll
  for (int i = 0; i < 4; ++i)
#pragma unroll
    for (int j = 0; j < 2; ++j) acc[i][j] = zero;

  // per-thread fixed LDS destination + source row for the async A-tile copy
  const int aidx = tid * 8;                      // 8 halves = 16 bytes per thread
  const int arow = aidx >> 5, akk = aidx & 31;
  const unsigned lds_a = (unsigned)(size_t)&As[aidx];   // LDS aperture: low 32 bits = LDS byte addr

  for (int k0 = 0; k0 < INTER; k0 += 32) {
    // A tile: f16 copy, global -> LDS without touching VGPR data path (ASYNCcnt)
    {
      const _Float16* gsrc = act + (size_t)(m0 + arow) * INTER + k0 + akk;
      asm volatile("global_load_async_to_lds_b128 %0, %1, off"
                   :
                   : "v"(lds_a), "v"(gsrc)
                   : "memory");
    }
    // B tile: thread owns output column `tid` (f32 -> f16 through VGPRs)
    {
      const float* cb = w2e + (n0 + tid);
#pragma unroll 4
      for (int kk = 0; kk < 32; ++kk)
        Bs[tid * 32 + kk] = (_Float16)cb[(size_t)(k0 + kk) * HIDDEN];
      if (k0 + 32 < INTER) {
#pragma unroll 4
        for (int kk = 0; kk < 32; kk += 8)
          __builtin_prefetch(&cb[(size_t)(k0 + 32 + kk) * HIDDEN], 0, 1);
      }
    }
    // wait for this wave's async LDS writes, then workgroup barrier
    asm volatile("s_wait_asynccnt 0x0" ::: "memory");
    __syncthreads();
#pragma unroll
    for (int mi = 0; mi < 4; ++mi) {
      const int row = mi * 16 + l15;
      v8h alo = *(const v8h*)&As[row * 32 + lh * 16];
      v8h ahi = *(const v8h*)&As[row * 32 + lh * 16 + 8];
      v16h a = __builtin_shufflevector(alo, ahi, 0,1,2,3,4,5,6,7,8,9,10,11,12,13,14,15);
#pragma unroll
      for (int ni = 0; ni < 2; ++ni) {
        const int col = wv * 32 + ni * 16 + l15;
        v8h bl = *(const v8h*)&Bs[col * 32 + lh * 16];
        v8h bh = *(const v8h*)&Bs[col * 32 + lh * 16 + 8];
        v16h b = __builtin_shufflevector(bl, bh, 0,1,2,3,4,5,6,7,8,9,10,11,12,13,14,15);
        acc[mi][ni] = __builtin_amdgcn_wmma_f32_16x16x32_f16(
            false, a, false, b, (short)0, acc[mi][ni], false, false);
      }
    }
    __syncthreads();
  }

#pragma unroll
  for (int mi = 0; mi < 4; ++mi)
#pragma unroll
    for (int r = 0; r < 8; ++r) {
      int row = m0 + mi * 16 + lh * 8 + r;
      float wt = pwt[row];
#pragma unroll
      for (int ni = 0; ni < 2; ++ni) {
        int col = n0 + wv * 32 + ni * 16 + l15;
        outp[(size_t)row * HIDDEN + col] = wt * acc[mi][ni][r];
      }
    }
}

// ---------------- combine the two expert partials per token ----------------
__global__ void moe_combine_kernel(const float* __restrict__ outp, const int* __restrict__ posmap,
                                   float* __restrict__ out, int T) {
  int idx = blockIdx.x * blockDim.x + threadIdx.x;
  int total = T * HIDDEN;
  if (idx >= total) return;
  int t = idx / HIDDEN, h = idx - t * HIDDEN;
  int p0 = posmap[2 * t], p1 = posmap[2 * t + 1];
  out[idx] = outp[(size_t)p0 * HIDDEN + h] + outp[(size_t)p1 * HIDDEN + h];
}

extern "C" void kernel_launch(void* const* d_in, const int* in_sizes, int n_in,
                              void* d_out, int out_size, void* d_ws, size_t ws_size,
                              hipStream_t stream) {
  (void)n_in; (void)out_size; (void)ws_size;
  const float* x  = (const float*)d_in[0];
  const float* gw = (const float*)d_in[1];
  const float* w1 = (const float*)d_in[2];
  const float* w2 = (const float*)d_in[3];
  float* out = (float*)d_out;

  const int T = in_sizes[0] / HIDDEN;          // 8192 tokens
  const int Pmax = 2 * T + NEXP * TM;          // padded pair capacity

  char* wsb = (char*)d_ws;
  size_t off = 0;
  auto alloc = [&](size_t bytes, size_t align) -> void* {
    off = (off + align - 1) & ~(align - 1);
    void* p = wsb + off; off += bytes; return p;
  };
  int*      ecount  = (int*)alloc(NEXP * 4, 16);
  int*      ecursor = (int*)alloc(NEXP * 4, 16);
  int*      poff    = (int*)alloc((NEXP + 1) * 4, 16);
  int*      t2i     = (int*)alloc((size_t)T * 2 * 4, 16);
  float*    t2w     = (float*)alloc((size_t)T * 2 * 4, 16);
  int*      posmap  = (int*)alloc((size_t)T * 2 * 4, 16);
  int*      ptok    = (int*)alloc((size_t)Pmax * 4, 16);
  float*    pwt     = (float*)alloc((size_t)Pmax * 4, 16);
  _Float16* act     = (_Float16*)alloc((size_t)Pmax * INTER * 2, 256);
  float*    outp    = (float*)alloc((size_t)Pmax * HIDDEN * 4, 256);

  moe_init_kernel<<<(Pmax + 255) / 256, 256, 0, stream>>>(ecount, ptok, pwt, Pmax);
  moe_gate_kernel<<<(T + 7) / 8, 256, 0, stream>>>(x, gw, t2i, t2w, ecount, T);
  moe_scan_kernel<<<1, 32, 0, stream>>>(ecount, poff, ecursor);
  moe_scatter_kernel<<<(T + 255) / 256, 256, 0, stream>>>(t2i, t2w, ecursor, ptok, pwt, posmap, T);

  const int mtiles = Pmax / TM;
  dim3 g1(mtiles, INTER / TN);
  moe_gemm1_kernel<<<g1, 256, 0, stream>>>(x, w1, ptok, poff, act);
  dim3 g2(mtiles, HIDDEN / TN);
  moe_gemm2_kernel<<<g2, 256, 0, stream>>>(act, w2, poff, pwt, outp);
  moe_combine_kernel<<<((T * HIDDEN) + 255) / 256, 256, 0, stream>>>(outp, posmap, out, T);
}
